// GATNet_6622839570675
// MI455X (gfx1250) — matrix operand order repaired and checked by reading, not compiled, not run on hardware
//
#include <hip/hip_runtime.h>
#include <hip/hip_bf16.h>

// ---------------------------------------------------------------------------
// GCN inference pipeline for MI455X (gfx1250, wave32, WMMA).
// GEMMs: v_wmma_f32_16x16x32_bf16, 128x128 block tile, 8 waves of 32x64.
// fp32 -> bf16 conversion happens during LDS staging (float4 + cvt_pk path
// for full tiles, guarded scalar path only for ragged edge tiles).
// ---------------------------------------------------------------------------

typedef __attribute__((ext_vector_type(16))) __bf16 v16bf;
typedef __attribute__((ext_vector_type(8)))  __bf16 v8bf;
typedef __attribute__((ext_vector_type(4)))  __bf16 v4bf;
typedef __attribute__((ext_vector_type(8)))  float  v8f;

#define TM 128
#define TN 128
#define TK 32

// Activation codes: 0 = none, 1 = relu, 2 = shrink: max(z - thr, 0)
__global__ __launch_bounds__(256) void gemm_bf16_wmma(
    const float* __restrict__ A,    // [M,K] row-major fp32
    const float* __restrict__ B,    // [K,N] row-major fp32
    const float* __restrict__ bias, // [N] or nullptr
    float* __restrict__ C,          // [M,N]
    int M, int N, int K, int act, const float* __restrict__ thr_p)
{
    __shared__ __align__(64) __bf16 As[TM][TK];  // row-major (8 KB)
    __shared__ __align__(64) __bf16 Bs[TN][TK];  // N-major   (8 KB)

    const int tid  = threadIdx.x;
    const int wave = tid >> 5;
    const int lane = tid & 31;
    const int m0 = blockIdx.y * TM;
    const int n0 = blockIdx.x * TN;
    const int wm = (wave >> 1) * 32;   // 0/32/64/96
    const int wn = (wave & 1) * 64;    // 0/64
    const int lh = lane & 15;
    const bool hiHalf = lane >= 16;

    v8f acc[2][4] = {};                 // [mi][ni] 16x16 f32 tiles

    const bool fullM = (m0 + TM <= M);
    const bool fullN = (n0 + TN <= N);

    const int ksteps = (K + TK - 1) / TK;
    for (int kt = 0; kt < ksteps; ++kt) {
        const int k0 = kt * TK;
        const bool fullK = (k0 + TK <= K);

        // ---- Stage A tile (128x32): 4096 elems, 16/thread --------------
        if (fullM & fullK) {
            #pragma unroll
            for (int u = tid; u < TM * (TK / 4); u += 256) {   // 4 units/thread
                int r = u >> 3, c4 = (u & 7) * 4;
                const float4 v = *(const float4*)&A[(size_t)(m0 + r) * K + k0 + c4];
                v4bf b = {(__bf16)v.x, (__bf16)v.y, (__bf16)v.z, (__bf16)v.w};
                *(v4bf*)&As[r][c4] = b;
            }
        } else {
            for (int i = tid; i < TM * TK; i += 256) {
                int r = i >> 5, c = i & (TK - 1);
                int gr = m0 + r, gc = k0 + c;
                As[r][c] = (__bf16)((gr < M && gc < K) ? A[(size_t)gr * K + gc] : 0.0f);
            }
        }
        // ---- Stage B tile (32x128 -> N-major [128][32]) ----------------
        if (fullN & fullK) {
            #pragma unroll
            for (int u = tid; u < (TK / 4) * TN; u += 256) {   // 4 units/thread
                int n = u & (TN - 1);
                int kq = (u >> 7) * 4;                          // 0,4,...,28
                const float* bp = &B[(size_t)(k0 + kq) * N + n0 + n];
                float f0 = bp[0];
                float f1 = bp[(size_t)N];
                float f2 = bp[(size_t)2 * N];
                float f3 = bp[(size_t)3 * N];
                v4bf b = {(__bf16)f0, (__bf16)f1, (__bf16)f2, (__bf16)f3};
                *(v4bf*)&Bs[n][kq] = b;
            }
        } else {
            for (int i = tid; i < TN * TK; i += 256) {
                int c = i >> 7;        // k 0..31
                int r = i & (TN - 1);  // n 0..127
                int gk = k0 + c, gn = n0 + r;
                Bs[r][c] = (__bf16)((gk < K && gn < N) ? B[(size_t)gk * N + gn] : 0.0f);
            }
        }
        __syncthreads();

        // A fragment (ISA 7.12.2 bf16 16x32): lane<16: K=0..7,16..23 ; lane>=16: K=8..15,24..31
        const int akb = hiHalf ? 8 : 0;
        v16bf afrag[2], bfrag[4];
        #pragma unroll
        for (int mi = 0; mi < 2; ++mi) {
            const __bf16* ap = &As[wm + mi * 16 + lh][0];
            v8bf lo = *(const v8bf*)(ap + akb);
            v8bf hi = *(const v8bf*)(ap + akb + 16);
            afrag[mi] = __builtin_shufflevector(lo, hi,
                0, 1, 2, 3, 4, 5, 6, 7, 8, 9, 10, 11, 12, 13, 14, 15);
        }
        // B fragment: lane<16: K=0..7,16..23 of column N=lane ; lane>=16: K=8..15,24..31
        const int bkb = hiHalf ? 8 : 0;
        #pragma unroll
        for (int ni = 0; ni < 4; ++ni) {
            const __bf16* bp = &Bs[wn + ni * 16 + lh][0];
            v8bf lo = *(const v8bf*)(bp + bkb);
            v8bf hi = *(const v8bf*)(bp + bkb + 16);
            bfrag[ni] = __builtin_shufflevector(lo, hi,
                0, 1, 2, 3, 4, 5, 6, 7, 8, 9, 10, 11, 12, 13, 14, 15);
        }

        #pragma unroll
        for (int mi = 0; mi < 2; ++mi)
            #pragma unroll
            for (int ni = 0; ni < 4; ++ni)
                acc[mi][ni] = __builtin_amdgcn_wmma_f32_16x16x32_bf16(
                    false, afrag[mi], false, bfrag[ni],
                    (short)0, acc[mi][ni], false, false);
        __syncthreads();
    }

    const float thr = (act == 2) ? *thr_p : 0.0f;
    // C/D layout: VGPR r -> row = r + (lane>=16 ? 8 : 0); col = lane&15.
    #pragma unroll
    for (int mi = 0; mi < 2; ++mi) {
        #pragma unroll
        for (int ni = 0; ni < 4; ++ni) {
            int col = n0 + wn + ni * 16 + lh;
            if (col >= N) continue;
            float bv = bias ? bias[col] : 0.0f;
            #pragma unroll
            for (int r = 0; r < 8; ++r) {
                int row = m0 + wm + mi * 16 + (hiHalf ? 8 : 0) + r;
                if (row >= M) continue;
                float z = acc[mi][ni][r] + bv;
                if (act == 1)      z = fmaxf(z, 0.0f);
                else if (act == 2) z = fmaxf(z - thr, 0.0f);
                C[(size_t)row * N + col] = z;
            }
        }
    }
}

// ---------------------------------------------------------------------------
__global__ __launch_bounds__(256) void zero_f32(float* p, int n) {
    int i = blockIdx.x * 256 + threadIdx.x;
    if (i < n) p[i] = 0.0f;
}

__global__ __launch_bounds__(256) void degree_kernel(const int* __restrict__ dst,
                                                     float* __restrict__ deg, int E) {
    int i = blockIdx.x * 256 + threadIdx.x;
    if (i < E) atomicAdd(&deg[dst[i]], 1.0f);
}

// dinv = rsqrt(indegree + 1)  (the +1 is the self-loop; degree >= 1 always)
__global__ __launch_bounds__(256) void dinv_kernel(const float* __restrict__ deg,
                                                   float* __restrict__ dinv, int n) {
    int i = blockIdx.x * 256 + threadIdx.x;
    if (i < n) dinv[i] = rsqrtf(deg[i] + 1.0f);
}

// agg[n][f] = hW[n][f] * dinv[n]^2   (self-loop contribution; also initializes agg)
__global__ __launch_bounds__(256) void init_agg_kernel(const float* __restrict__ hW,
                                                       const float* __restrict__ dinv,
                                                       float* __restrict__ agg,
                                                       int n, int F) {
    int i = blockIdx.x * 256 + threadIdx.x;
    if (i >= n * F) return;
    float d = dinv[i / F];
    agg[i] = hW[i] * d * d;
}

// One thread per (edge, 4-float feature chunk): agg[dst] += hW[src] * dinv[s]*dinv[d]
__global__ __launch_bounds__(256) void scatter_kernel(const int* __restrict__ src,
                                                      const int* __restrict__ dst,
                                                      const float* __restrict__ dinv,
                                                      const float* __restrict__ hW,
                                                      float* __restrict__ agg,
                                                      int E, int F) {
    const int chunks = F >> 2;
    int idx = blockIdx.x * 256 + threadIdx.x;
    if (idx >= E * chunks) return;
    int e = idx / chunks;
    int c = (idx - e * chunks) << 2;
    int s = src[e], d = dst[e];
    float w = dinv[s] * dinv[d];
    const float4 v = *(const float4*)&hW[(size_t)s * F + c];
    float* o = &agg[(size_t)d * F + c];
    atomicAdd(o + 0, v.x * w);
    atomicAdd(o + 1, v.y * w);
    atomicAdd(o + 2, v.z * w);
    atomicAdd(o + 3, v.w * w);
}

// h = elu(agg + bias)
__global__ __launch_bounds__(256) void finalize_elu_kernel(const float* __restrict__ agg,
                                                           const float* __restrict__ bias,
                                                           float* __restrict__ h,
                                                           int n, int F) {
    int i = blockIdx.x * 256 + threadIdx.x;
    if (i >= n * F) return;
    float z = agg[i] + bias[i % F];
    h[i] = (z > 0.0f) ? z : (expf(z) - 1.0f);
}

// One block per pair: xc = normalize(concat(h[i1], h[i2]))  (F = 128, block = 256)
__global__ __launch_bounds__(256) void pair_gather_norm_kernel(const float* __restrict__ h,
                                                               const int* __restrict__ i1,
                                                               const int* __restrict__ i2,
                                                               float* __restrict__ xc,
                                                               int F) {
    __shared__ float red[256];
    int p = blockIdx.x, t = threadIdx.x;
    float v = (t < F) ? h[(size_t)i1[p] * F + t]
                      : h[(size_t)i2[p] * F + (t - F)];
    red[t] = v * v;
    __syncthreads();
    for (int s = 128; s > 0; s >>= 1) {
        if (t < s) red[t] += red[t + s];
        __syncthreads();
    }
    float nrm = fmaxf(sqrtf(red[0]), 1e-12f);
    xc[(size_t)p * (2 * F) + t] = v / nrm;
}

// ---------------------------------------------------------------------------
extern "C" void kernel_launch(void* const* d_in, const int* in_sizes, int n_in,
                              void* d_out, int out_size, void* d_ws, size_t ws_size,
                              hipStream_t stream) {
    constexpr int NN = 30000, NE = 480000, NP = 4096;
    constexpr size_t MiB = 1ull << 20;

    const float* cid   = (const float*)d_in[0];
    const int*   edges = (const int*)d_in[1];
    const int*   e_src = edges;
    const int*   e_dst = edges + NE;
    const int*   i1    = (const int*)d_in[2];
    const int*   i2    = (const int*)d_in[3];
    const float* thr   = (const float*)d_in[4];
    const float *bb2W = (const float*)d_in[5],  *bb2b = (const float*)d_in[6];
    const float *bb3W = (const float*)d_in[7],  *bb3b = (const float*)d_in[8];
    const float* gW[5] = {(const float*)d_in[9],  (const float*)d_in[11],
                          (const float*)d_in[13], (const float*)d_in[15],
                          (const float*)d_in[17]};
    const float* gb[5] = {(const float*)d_in[10], (const float*)d_in[12],
                          (const float*)d_in[14], (const float*)d_in[16],
                          (const float*)d_in[18]};
    const float *fc1W = (const float*)d_in[19], *fc1b = (const float*)d_in[20];
    const float *fc2W = (const float*)d_in[21], *fc2b = (const float*)d_in[22];
    const float *fc3W = (const float*)d_in[23], *fc3b = (const float*)d_in[24];
    const float *outW = (const float*)d_in[25], *outb = (const float*)d_in[26];
    float* out = (float*)d_out;

    // Workspace layout (peak ~198 MiB). x1 region (0..~101 MiB) is reused for
    // hW/agg once GEMM2 has consumed x1.
    char* ws = (char*)d_ws;
    float* x1   = (float*)(ws + 0);         // [30000, 881]  fp32
    float* hW   = (float*)(ws + 0);         // [30000, 256]  (reuse of x1 region)
    float* agg  = (float*)(ws + 32  * MiB); // [30000, 256]
    float* hA   = (float*)(ws + 104 * MiB); // [30000, 256]
    float* hB   = (float*)(ws + 136 * MiB); // [30000, 256]
    float* deg  = (float*)(ws + 168 * MiB); // [30000]
    float* dinv = (float*)(ws + 169 * MiB); // [30000]
    float* xc   = (float*)(ws + 170 * MiB); // [4096, 256]
    float* y1   = (float*)(ws + 175 * MiB); // [4096, 1024]
    float* y2   = (float*)(ws + 192 * MiB); // [4096, 256]
    float* y3   = (float*)(ws + 197 * MiB); // [4096, 64]

    auto gemm = [&](const float* A, const float* B, const float* bias, float* C,
                    int M, int N, int K, int act) {
        dim3 grid((N + TN - 1) / TN, (M + TM - 1) / TM);
        gemm_bf16_wmma<<<grid, 256, 0, stream>>>(A, B, bias, C, M, N, K, act, thr);
    };
    auto blocks = [](int n) { return (n + 255) / 256; };

    // --- BasicBlock: relu(cid@W2+b2) then soft-threshold == max(z - thr, 0) ---
    gemm(cid, bb2W, bb2b, x1, NN, 881, 881, /*shrink*/2);
    // relu(x1 @ W3 + b3) -> hA  [30000, 256]
    gemm(x1, bb3W, bb3b, hA, NN, 256, 881, /*relu*/1);

    // --- GCN symmetric normalization ---
    zero_f32<<<blocks(NN), 256, 0, stream>>>(deg, NN);
    degree_kernel<<<blocks(NE), 256, 0, stream>>>(e_dst, deg, NE);
    dinv_kernel<<<blocks(NN), 256, 0, stream>>>(deg, dinv, NN);

    // --- 5 GCN layers ---
    const int fin[5]  = {256, 256, 256, 256, 128};
    const int fout[5] = {256, 256, 256, 128, 128};
    float* hin = hA;
    float* hout = hB;
    for (int l = 0; l < 5; ++l) {
        const int F = fout[l];
        gemm(hin, gW[l], nullptr, hW, NN, F, fin[l], /*none*/0);
        init_agg_kernel<<<blocks(NN * F), 256, 0, stream>>>(hW, dinv, agg, NN, F);
        scatter_kernel<<<blocks(NE * (F >> 2)), 256, 0, stream>>>(
            e_src, e_dst, dinv, hW, agg, NE, F);
        finalize_elu_kernel<<<blocks(NN * F), 256, 0, stream>>>(agg, gb[l], hout, NN, F);
        float* t = hin; hin = hout; hout = t;
    }
    // final node embeddings (128 feats) now in hin

    // --- pair readout ---
    pair_gather_norm_kernel<<<NP, 256, 0, stream>>>(hin, i1, i2, xc, 128);
    gemm(xc, fc1W, fc1b, y1, NP, 1024, 256, 1);
    gemm(y1, fc2W, fc2b, y2, NP, 256, 1024, 1);
    gemm(y2, fc3W, fc3b, y3, NP, 64, 256, 1);
    gemm(y3, outW, outb, out, NP, 2, 64, 0);
}
